// VIT_30537217475341
// MI455X (gfx1250) — compile-verified, matrix operand order checked by reading, not run on hardware
//
#include <hip/hip_runtime.h>
#include <hip/hip_bf16.h>
#include <math.h>

// ---------------------------------------------------------------------------
// ViT-Base forward for MI455X (gfx1250). All GEMMs run through
// v_wmma_f32_16x16x32_f16 (wave32 WMMA); A tiles are streamed into LDS with
// the CDNA5 async-to-LDS path (ASYNCcnt) and double-buffered so global
// traffic overlaps the WMMA pipeline. fp32 params are converted to fp16 on
// the fly; accumulation is fp32.
// ---------------------------------------------------------------------------

typedef __attribute__((ext_vector_type(16))) _Float16 v16h;
typedef __attribute__((ext_vector_type(8)))  _Float16 v8h;
typedef __attribute__((ext_vector_type(8)))  float    v8f;

#define VB   32
#define VS   197
#define VHD  768
#define VNH  12
#define VDH  64
#define VDFF 3072
#define VL   12
#define VOUT 1000
#define NPAT 196           // 14*14 patches

__device__ __forceinline__ void wait_async0()
{
#if __has_builtin(__builtin_amdgcn_s_wait_asynccnt)
    __builtin_amdgcn_s_wait_asynccnt(0);
#else
    asm volatile("s_wait_asynccnt 0x0" ::: "memory");
#endif
}

// Issue two async 16B global->LDS copies covering 16 fp16 (half of a 32-wide K-row chunk).
// The 24-bit instruction offset applies to both LDS and global addresses
// (ISA 15.18.3), so one base pair serves both halves.
__device__ __forceinline__ void async_copy_32h(const _Float16* gsrc, void* lds)
{
    const unsigned long long g = (unsigned long long)(size_t)gsrc;
    const unsigned           l = (unsigned)(size_t)lds;   // flat low32 == LDS offset
    asm volatile(
        "global_load_async_to_lds_b128 %0, %1, off\n\t"
        "global_load_async_to_lds_b128 %0, %1, off offset:16"
        :: "v"(l), "v"(g) : "memory");
}

// ---------------------------------------------------------------------------
// Generic fp16 WMMA GEMM: C = epilogue(A[M,K] * B[K,N] + bias[N])
// Block = 256 threads (8 waves). Tile 128x128, K-step 32, double-buffered LDS.
// Wave grid 4(M) x 2(N); each wave owns a 32x64 sub-tile (2x4 WMMA accums).
// A tiles: async global->LDS (no VGPR round trip). B tiles: register-staged
// (transpose to [n][k]) and software-pipelined across the WMMA block.
// EPI: 0 = fp32 store, 1 = fp32 in-place residual add, 2 = GELU -> fp16,
//      3 = fp16 store.
// NOTE: B must tolerate reads up to the 128-aligned tile edge (true for the
// fp16 weight staging buffer); A rows >= M are never loaded and their (stale)
// LDS contents only feed C rows the epilogue discards.
// ---------------------------------------------------------------------------
template<int EPI>
__launch_bounds__(256)
__global__ void gemm_wmma(const _Float16* __restrict__ A,
                          const _Float16* __restrict__ B,
                          const float* __restrict__ bias,
                          float* __restrict__ outF,
                          _Float16* __restrict__ outH,
                          int M, int N, int K)
{
    // K-contiguous fp16 tiles; row stride 40 halves (80B) keeps 16B alignment
    __shared__ __align__(16) _Float16 As[2][128][40];
    __shared__ __align__(16) _Float16 Bs[2][128][40];   // transposed: [n][k]

    const int tid   = threadIdx.x;
    const int lane  = tid & 31;
    const int wid   = tid >> 5;
    const int waveM = wid & 3;     // 0..3  -> 32 rows each
    const int waveN = wid >> 2;    // 0..1  -> 64 cols each
    const int hi    = lane >> 4;   // 0/1 half-wave (K-group select)
    const int lm    = lane & 15;
    const int mb    = blockIdx.y * 128;
    const int nb    = blockIdx.x * 128;

    v8f acc[2][4] = {};

    const int aRow = tid >> 1;            // 0..127
    const int aCol = (tid & 1) << 4;      // 0 or 16 (halves)
    const int bK   = tid >> 3;            // 0..31
    const int bN   = (tid & 7) << 4;      // 0..112 (halves)
    const int gmA  = mb + aRow;

    union BStage { uint4 u[2]; _Float16 h[16]; } bst;

    // ---- prologue: stage tile 0 ----
    if (gmA < M)
        async_copy_32h(A + (size_t)gmA * K + aCol, &As[0][aRow][aCol]);
    {
        const uint4* src =
            reinterpret_cast<const uint4*>(B + (size_t)bK * N + nb + bN);
        bst.u[0] = src[0];
        bst.u[1] = src[1];
#pragma unroll
        for (int i = 0; i < 16; ++i) Bs[0][bN + i][bK] = bst.h[i];
    }
    wait_async0();
    __syncthreads();

    const int NKT = K >> 5;
    for (int it = 0; it < NKT; ++it) {
        const int  cur  = it & 1;
        const int  nxt  = cur ^ 1;
        const bool more = (it + 1) < NKT;
        const int  ktn  = (it + 1) << 5;

        // ---- prefetch tile it+1 while tile it computes ----
        if (more) {
            if (gmA < M)
                async_copy_32h(A + (size_t)gmA * K + ktn + aCol,
                               &As[nxt][aRow][aCol]);
            const uint4* src = reinterpret_cast<const uint4*>(
                B + (size_t)(ktn + bK) * N + nb + bN);
            bst.u[0] = src[0];
            bst.u[1] = src[1];
        }

        // ---- fragment loads (2x16B LDS reads each, ISA 16-bit A layout) ----
        v16h aF[2], bF[4];
#pragma unroll
        for (int mi = 0; mi < 2; ++mi) {
            const int r = waveM * 32 + mi * 16 + lm;
            v8h x0 = *reinterpret_cast<const v8h*>(&As[cur][r][hi * 8]);
            v8h x1 = *reinterpret_cast<const v8h*>(&As[cur][r][16 + hi * 8]);
            aF[mi] = __builtin_shufflevector(x0, x1, 0, 1, 2, 3, 4, 5, 6, 7,
                                             8, 9, 10, 11, 12, 13, 14, 15);
        }
#pragma unroll
        for (int ni = 0; ni < 4; ++ni) {
            const int rc = waveN * 64 + ni * 16 + lm;
            v8h x0 = *reinterpret_cast<const v8h*>(&Bs[cur][rc][hi * 8]);
            v8h x1 = *reinterpret_cast<const v8h*>(&Bs[cur][rc][16 + hi * 8]);
            bF[ni] = __builtin_shufflevector(x0, x1, 0, 1, 2, 3, 4, 5, 6, 7,
                                             8, 9, 10, 11, 12, 13, 14, 15);
        }

#pragma unroll
        for (int mi = 0; mi < 2; ++mi)
#pragma unroll
            for (int ni = 0; ni < 4; ++ni)
                acc[mi][ni] = __builtin_amdgcn_wmma_f32_16x16x32_f16(
                    false, aF[mi], false, bF[ni], (short)0, acc[mi][ni],
                    false, false);

        // ---- land the staged B tile into the alternate buffer ----
        if (more) {
#pragma unroll
            for (int i = 0; i < 16; ++i) Bs[nxt][bN + i][bK] = bst.h[i];
        }
        wait_async0();
        __syncthreads();
    }

    // ---- epilogue (C layout: lane 0-15 -> M=r, lane 16-31 -> M=r+8) ----
#pragma unroll
    for (int mi = 0; mi < 2; ++mi) {
#pragma unroll
        for (int ni = 0; ni < 4; ++ni) {
            const int gm0 = mb + waveM * 32 + mi * 16 + hi * 8;
            const int gn  = nb + waveN * 64 + ni * 16 + lm;
            if (gn >= N) continue;
            const float bvl = bias[gn];
#pragma unroll
            for (int r = 0; r < 8; ++r) {
                const int gm = gm0 + r;
                if (gm >= M) continue;
                const float v = acc[mi][ni][r] + bvl;
                const size_t idx = (size_t)gm * N + gn;
                if (EPI == 0) {
                    outF[idx] = v;
                } else if (EPI == 1) {
                    outF[idx] += v;          // in-place residual
                } else if (EPI == 2) {
                    const float g = 0.5f * v * (1.0f + erff(v * 0.70710678118f));
                    outH[idx] = (_Float16)g;
                } else {
                    outH[idx] = (_Float16)v;
                }
            }
        }
    }
}

// ---------------------------------------------------------------------------
// Attention: one block per (batch, head). K/V head tiles cached in LDS.
// Each of 8 waves processes query rows w, w+8, ... (25 uniform iterations).
// ---------------------------------------------------------------------------
__launch_bounds__(256)
__global__ void attention_k(const _Float16* __restrict__ q,
                            const _Float16* __restrict__ kmat,
                            const _Float16* __restrict__ vmat,
                            _Float16* __restrict__ o)
{
    __shared__ __align__(16) _Float16 Ks[VS * VDH];
    __shared__ __align__(16) _Float16 Vs[VS * VDH];
    __shared__ float    attBuf[8 * VS];
    __shared__ _Float16 qs[8 * VDH];

    const int tid  = threadIdx.x;
    const int lane = tid & 31;
    const int w    = tid >> 5;
    const int b    = blockIdx.x / VNH;
    const int h    = blockIdx.x % VNH;
    const size_t base = ((size_t)b * VS) * VHD + (size_t)h * VDH;

    // load K and V head tiles (16B chunks)
    for (int c = tid; c < (VS * VDH) / 8; c += 256) {
        const int off = c * 8;
        const int row = off >> 6;
        const int col = off & 63;
        *reinterpret_cast<uint4*>(&Ks[off]) =
            *reinterpret_cast<const uint4*>(kmat + base + (size_t)row * VHD + col);
        *reinterpret_cast<uint4*>(&Vs[off]) =
            *reinterpret_cast<const uint4*>(vmat + base + (size_t)row * VHD + col);
    }
    __syncthreads();

    for (int it = 0; it < 25; ++it) {
        const int  qIdx = it * 8 + w;
        const bool act  = qIdx < VS;
        if (act) {
            qs[w * VDH + lane]      = q[base + (size_t)qIdx * VHD + lane];
            qs[w * VDH + lane + 32] = q[base + (size_t)qIdx * VHD + lane + 32];
        }
        __syncthreads();

        if (act) {
            float sc[7];
#pragma unroll
            for (int j = 0; j < 7; ++j) {
                const int kk = lane + 32 * j;
                float d = 0.0f;
                if (kk < VS) {
                    const _Float16* kr = &Ks[kk * VDH];
                    const _Float16* qr = &qs[w * VDH];
                    for (int dd = 0; dd < VDH; ++dd)
                        d += (float)qr[dd] * (float)kr[dd];
                }
                sc[j] = (kk < VS) ? d * 0.125f : -3.0e38f;
            }
            float mx = sc[0];
#pragma unroll
            for (int j = 1; j < 7; ++j) mx = fmaxf(mx, sc[j]);
#pragma unroll
            for (int off = 16; off > 0; off >>= 1)
                mx = fmaxf(mx, __shfl_xor(mx, off, 32));
            float sum = 0.0f;
#pragma unroll
            for (int j = 0; j < 7; ++j) {
                const float e = (sc[j] > -1.0e38f) ? __expf(sc[j] - mx) : 0.0f;
                sc[j] = e;
                sum += e;
            }
#pragma unroll
            for (int off = 16; off > 0; off >>= 1)
                sum += __shfl_xor(sum, off, 32);
            const float inv = 1.0f / sum;
#pragma unroll
            for (int j = 0; j < 7; ++j) {
                const int kk = lane + 32 * j;
                if (kk < VS) attBuf[w * VS + kk] = sc[j] * inv;
            }
        }
        __syncthreads();

        if (act) {
            float o0 = 0.0f, o1 = 0.0f;
            const float* ab = &attBuf[w * VS];
            for (int kk = 0; kk < VS; ++kk) {
                const float a = ab[kk];
                o0 += a * (float)Vs[kk * VDH + lane];
                o1 += a * (float)Vs[kk * VDH + 32 + lane];
            }
            o[base + (size_t)qIdx * VHD + lane]      = (_Float16)o0;
            o[base + (size_t)qIdx * VHD + lane + 32] = (_Float16)o1;
        }
        __syncthreads();
    }
}

// ---------------------------------------------------------------------------
// LayerNorm over HD=768, fp32 in -> fp16 out. One block per token row.
// ---------------------------------------------------------------------------
__launch_bounds__(256)
__global__ void layernorm_k(const float* __restrict__ x,
                            const float* __restrict__ g,
                            const float* __restrict__ bta,
                            _Float16* __restrict__ out)
{
    __shared__ float s1[256], s2[256];
    const int row = blockIdx.x, tid = threadIdx.x;
    const float* xr = x + (size_t)row * VHD;
    const float a0 = xr[tid], a1 = xr[tid + 256], a2 = xr[tid + 512];
    s1[tid] = a0 + a1 + a2;
    s2[tid] = a0 * a0 + a1 * a1 + a2 * a2;
    __syncthreads();
    for (int st = 128; st > 0; st >>= 1) {
        if (tid < st) { s1[tid] += s1[tid + st]; s2[tid] += s2[tid + st]; }
        __syncthreads();
    }
    const float mean = s1[0] * (1.0f / 768.0f);
    const float var  = s2[0] * (1.0f / 768.0f) - mean * mean;
    const float rstd = rsqrtf(var + 1e-5f);
    _Float16* orow = out + (size_t)row * VHD;
    orow[tid]       = (_Float16)((a0 - mean) * rstd * g[tid]       + bta[tid]);
    orow[tid + 256] = (_Float16)((a1 - mean) * rstd * g[tid + 256] + bta[tid + 256]);
    orow[tid + 512] = (_Float16)((a2 - mean) * rstd * g[tid + 512] + bta[tid + 512]);
}

// ---------------------------------------------------------------------------
// Small elementwise kernels
// ---------------------------------------------------------------------------
__global__ void convert_k(const float* __restrict__ in,
                          _Float16* __restrict__ out, size_t n)
{
    size_t i  = (size_t)blockIdx.x * blockDim.x + threadIdx.x;
    size_t st = (size_t)gridDim.x * blockDim.x;
    for (; i < n; i += st) out[i] = (_Float16)in[i];
}

__global__ void patchify_k(const float* __restrict__ img,
                           _Float16* __restrict__ Xp)
{
    const int idx = blockIdx.x * 256 + threadIdx.x;
    if (idx >= VB * NPAT * VHD) return;
    const int row = idx / VHD, col = idx - row * VHD;
    const int b = row / NPAT, pp = row - b * NPAT;
    const int i = pp / 14, j = pp - i * 14;
    const int c = col >> 8, rem = col & 255, ph = rem >> 4, pw = rem & 15;
    Xp[idx] = (_Float16)img[(((size_t)b * 3 + c) * 224 + (i * 16 + ph)) * 224 +
                            (j * 16 + pw)];
}

__global__ void assemble_k(const float* __restrict__ tok,
                           const float* __restrict__ cls,
                           const float* __restrict__ pos,
                           float* __restrict__ x)
{
    const int idx = blockIdx.x * 256 + threadIdx.x;
    if (idx >= VB * VS * VHD) return;
    const int c  = idx % VHD;
    const int bt = idx / VHD;
    const int t  = bt % VS;
    const int b  = bt / VS;
    const float v = (t == 0) ? cls[c]
                             : tok[((size_t)b * NPAT + (t - 1)) * VHD + c];
    x[idx] = v + pos[(size_t)t * VHD + c];
}

__global__ void extract_cls_k(const float* __restrict__ x,
                              _Float16* __restrict__ cls16)
{
    const int idx = blockIdx.x * 256 + threadIdx.x;
    if (idx >= VB * VHD) return;
    const int b = idx / VHD, c = idx % VHD;
    cls16[idx] = (_Float16)x[((size_t)b * VS) * VHD + c];
}

__launch_bounds__(256)
__global__ void softmax_head_k(const float* __restrict__ logits,
                               float* __restrict__ out)
{
    __shared__ float red[256];
    const int row = blockIdx.x, tid = threadIdx.x;
    const float* lr = logits + (size_t)row * VOUT;
    float mx = -3.0e38f;
    for (int c = tid; c < VOUT; c += 256) mx = fmaxf(mx, lr[c]);
    red[tid] = mx;
    __syncthreads();
    for (int st = 128; st > 0; st >>= 1) {
        if (tid < st) red[tid] = fmaxf(red[tid], red[tid + st]);
        __syncthreads();
    }
    const float m = red[0];
    __syncthreads();
    float s = 0.0f;
    for (int c = tid; c < VOUT; c += 256) {
        const float e = __expf(lr[c] - m);
        out[(size_t)row * VOUT + c] = e;
        s += e;
    }
    red[tid] = s;
    __syncthreads();
    for (int st = 128; st > 0; st >>= 1) {
        if (tid < st) red[tid] += red[tid + st];
        __syncthreads();
    }
    const float inv = 1.0f / red[0];
    for (int c = tid; c < VOUT; c += 256)
        out[(size_t)row * VOUT + c] *= inv;
}

// ---------------------------------------------------------------------------
// Host orchestration
// ---------------------------------------------------------------------------
extern "C" void kernel_launch(void* const* d_in, const int* in_sizes, int n_in,
                              void* d_out, int out_size, void* d_ws,
                              size_t ws_size, hipStream_t stream)
{
    (void)in_sizes; (void)n_in; (void)out_size; (void)ws_size;
    const float* images = (const float*)d_in[0];
    const float* W_map  = (const float*)d_in[1];
    const float* b_map  = (const float*)d_in[2];
    const float* cls    = (const float*)d_in[3];
    const float* pos    = (const float*)d_in[4];
    const float* ln1_g  = (const float*)d_in[5];
    const float* ln1_b  = (const float*)d_in[6];
    const float* Wq     = (const float*)d_in[7];
    const float* bq     = (const float*)d_in[8];
    const float* Wk     = (const float*)d_in[9];
    const float* bk     = (const float*)d_in[10];
    const float* Wv     = (const float*)d_in[11];
    const float* bv     = (const float*)d_in[12];
    const float* Wo     = (const float*)d_in[13];
    const float* bo     = (const float*)d_in[14];
    const float* ln2_g  = (const float*)d_in[15];
    const float* ln2_b  = (const float*)d_in[16];
    const float* W1     = (const float*)d_in[17];
    const float* b1     = (const float*)d_in[18];
    const float* W2     = (const float*)d_in[19];
    const float* b2     = (const float*)d_in[20];
    const float* Wh     = (const float*)d_in[21];
    const float* bh     = (const float*)d_in[22];
    float* out = (float*)d_out;

    const int ROWS_P = VB * NPAT;   // 6272
    const int ROWS_T = VB * VS;     // 6304

    char* p = (char*)d_ws;
    auto alloc = [&](size_t bytes) -> char* {
        char* r = p;
        p += (bytes + 255) & ~(size_t)255;
        return r;
    };
    _Float16* Xp     = (_Float16*)alloc((size_t)ROWS_P * VHD * 2);
    float*    tok    = (float*)   alloc((size_t)ROWS_P * VHD * 4);
    float*    x      = (float*)   alloc((size_t)ROWS_T * VHD * 4);
    _Float16* hbuf   = (_Float16*)alloc((size_t)ROWS_T * VHD * 2);
    _Float16* qb     = (_Float16*)alloc((size_t)ROWS_T * VHD * 2);
    _Float16* kb     = (_Float16*)alloc((size_t)ROWS_T * VHD * 2);
    _Float16* vb     = (_Float16*)alloc((size_t)ROWS_T * VHD * 2);
    _Float16* ob     = (_Float16*)alloc((size_t)ROWS_T * VHD * 2);
    _Float16* hg     = (_Float16*)alloc((size_t)ROWS_T * VDFF * 2);
    _Float16* wst    = (_Float16*)alloc((size_t)VHD * VDFF * 2);  // weight stage
    _Float16* cls16  = (_Float16*)alloc((size_t)VB * VHD * 2);
    float*    logits = (float*)   alloc((size_t)VB * VOUT * 4);

    auto conv = [&](const float* src, _Float16* dst, size_t n) {
        convert_k<<<2048, 256, 0, stream>>>(src, dst, n);
    };
    auto ggrid = [](int M, int N) {
        return dim3((unsigned)((N + 127) / 128), (unsigned)((M + 127) / 128));
    };

    // 1. patchify -> fp16 patch matrix
    patchify_k<<<(ROWS_P * VHD + 255) / 256, 256, 0, stream>>>(images, Xp);

    // 2. patch embedding GEMM (fp32 out)
    conv(W_map, wst, (size_t)VHD * VHD);
    gemm_wmma<0><<<ggrid(ROWS_P, VHD), 256, 0, stream>>>(
        Xp, wst, b_map, tok, nullptr, ROWS_P, VHD, VHD);

    // 3. assemble x = concat(cls, tokens) + pos
    assemble_k<<<(ROWS_T * VHD + 255) / 256, 256, 0, stream>>>(tok, cls, pos, x);

    // 4. transformer layers
    for (int l = 0; l < VL; ++l) {
        const size_t oHH = (size_t)l * VHD * VHD;
        const size_t oHD = (size_t)l * VHD;
        const size_t oHF = (size_t)l * VHD * VDFF;
        const size_t oDF = (size_t)l * VDFF;

        layernorm_k<<<ROWS_T, 256, 0, stream>>>(x, ln1_g + oHD, ln1_b + oHD, hbuf);

        conv(Wq + oHH, wst, (size_t)VHD * VHD);
        gemm_wmma<3><<<ggrid(ROWS_T, VHD), 256, 0, stream>>>(
            hbuf, wst, bq + oHD, nullptr, qb, ROWS_T, VHD, VHD);
        conv(Wk + oHH, wst, (size_t)VHD * VHD);
        gemm_wmma<3><<<ggrid(ROWS_T, VHD), 256, 0, stream>>>(
            hbuf, wst, bk + oHD, nullptr, kb, ROWS_T, VHD, VHD);
        conv(Wv + oHH, wst, (size_t)VHD * VHD);
        gemm_wmma<3><<<ggrid(ROWS_T, VHD), 256, 0, stream>>>(
            hbuf, wst, bv + oHD, nullptr, vb, ROWS_T, VHD, VHD);

        attention_k<<<VB * VNH, 256, 0, stream>>>(qb, kb, vb, ob);

        conv(Wo + oHH, wst, (size_t)VHD * VHD);
        gemm_wmma<1><<<ggrid(ROWS_T, VHD), 256, 0, stream>>>(
            ob, wst, bo + oHD, x, nullptr, ROWS_T, VHD, VHD);

        layernorm_k<<<ROWS_T, 256, 0, stream>>>(x, ln2_g + oHD, ln2_b + oHD, hbuf);

        conv(W1 + oHF, wst, (size_t)VHD * VDFF);
        gemm_wmma<2><<<ggrid(ROWS_T, VDFF), 256, 0, stream>>>(
            hbuf, wst, b1 + oDF, nullptr, hg, ROWS_T, VDFF, VHD);
        conv(W2 + oHF, wst, (size_t)VHD * VDFF);
        gemm_wmma<1><<<ggrid(ROWS_T, VHD), 256, 0, stream>>>(
            hg, wst, b2 + oHD, x, nullptr, ROWS_T, VHD, VDFF);
    }

    // 5. classification head + softmax
    extract_cls_k<<<(VB * VHD + 255) / 256, 256, 0, stream>>>(x, cls16);
    conv(Wh, wst, (size_t)VHD * VOUT);
    gemm_wmma<0><<<ggrid(VB, VOUT), 256, 0, stream>>>(
        cls16, wst, bh, logits, nullptr, VB, VOUT, VHD);
    softmax_head_k<<<VB, 256, 0, stream>>>(logits, out);
}
